// DeltaModel_3204045603604
// MI455X (gfx1250) — matrix operand order repaired and checked by reading, not compiled, not run on hardware
//
#include <hip/hip_runtime.h>

typedef __bf16 bf16;
typedef __attribute__((ext_vector_type(16))) bf16 v16bf;
typedef __attribute__((ext_vector_type(8)))  bf16 v8bf;
typedef __attribute__((ext_vector_type(8)))  float v8f;
typedef unsigned int u32x4 __attribute__((ext_vector_type(4)));
typedef int i32x8 __attribute__((ext_vector_type(8)));
typedef int i32x4 __attribute__((ext_vector_type(4)));

#define HDIM 64
#define FDIM 128
#define SEQL 2048
#define NB   256
#define CH   64

// ---------- WMMA fragment helpers (CDNA5 wave32 layouts, CDNA5 ISA) ----------

__device__ __forceinline__ v16bf a_frag(const bf16* a, int lda) {
  const int lane = (int)(threadIdx.x & 31);
  const int row  = lane & 15;
  const int kb   = (lane >> 4) << 3;            // 0 or 8
  const bf16* p = a + row * lda + kb;
  v8bf lo = *(const v8bf*)(p);
  v8bf hi = *(const v8bf*)(p + 16);
  v16bf r;
#pragma unroll
  for (int i = 0; i < 8; ++i) { r[i] = lo[i]; r[8 + i] = hi[i]; }
  return r;
}

__device__ __forceinline__ v16bf bt_frag(const bf16* bt, int ldb) {
  const int lane = (int)(threadIdx.x & 31);
  const int col  = lane & 15;
  const int kb   = (lane >> 4) << 3;            // 0 or 8
  const bf16* p = bt + col * ldb + kb;
  v8bf lo = *(const v8bf*)(p);
  v8bf hi = *(const v8bf*)(p + 16);
  v16bf r;
#pragma unroll
  for (int i = 0; i < 8; ++i) { r[i] = lo[i]; r[8 + i] = hi[i]; }
  return r;
}

__device__ __forceinline__ v8f cd_load(const float* c, int ldc) {
  const int lane = (int)(threadIdx.x & 31);
  const int col  = lane & 15;
  const int rb   = (lane >> 4) << 3;
  v8f r;
#pragma unroll
  for (int i = 0; i < 8; ++i) r[i] = c[(rb + i) * ldc + col];
  return r;
}

__device__ __forceinline__ void cd_store(v8f acc, float* c, int ldc) {
  const int lane = (int)(threadIdx.x & 31);
  const int col  = lane & 15;
  const int rb   = (lane >> 4) << 3;
#pragma unroll
  for (int i = 0; i < 8; ++i) c[(rb + i) * ldc + col] = acc[i];
}

// D[M,N](f32 LDS, ldd) = A[M,K](bf16 LDS, lda) * Bt[N,K]^T(bf16 LDS, ldb) (+Cin)
__device__ void gemm_lds(float* D, int ldd, const bf16* A, int lda,
                         const bf16* Bt, int ldb, int M, int N, int K,
                         const float* Cin) {
  const int wave = (int)(threadIdx.x >> 5);
  const int nw   = (int)(blockDim.x >> 5);
  const int nTn  = N >> 4;
  const int nt   = (M >> 4) * nTn;
  for (int t = wave; t < nt; t += nw) {
    const int tm = t / nTn, tn = t - tm * nTn;
    v8f acc;
    if (Cin) {
      acc = cd_load(Cin + tm * 16 * ldd + tn * 16, ldd);
    } else {
#pragma unroll
      for (int i = 0; i < 8; ++i) acc[i] = 0.f;
    }
    for (int kk = 0; kk < K; kk += 32) {
      v16bf af  = a_frag(A + tm * 16 * lda + kk, lda);
      v16bf bf_ = bt_frag(Bt + tn * 16 * ldb + kk, ldb);
      acc = __builtin_amdgcn_wmma_f32_16x16x32_bf16(false, af, false, bf_,
                                                    (short)0, acc, false, false);
    }
    cd_store(acc, D + tm * 16 * ldd + tn * 16, ldd);
  }
}

// ---------- TDM descriptor builders (CDNA5 ISA) ----------

// Plain 2D tile load: tensor (dim1 rows x dim0 cols) of f32, tile == whole tensor.
__device__ __forceinline__ void tdm_load_2d(unsigned lds_off, const void* gptr,
                                            unsigned dim0, unsigned dim1) {
  unsigned long long ga = (unsigned long long)(uintptr_t)gptr;
  u32x4 g0;
  g0[0] = 1u;                                             // count=1, no gather
  g0[1] = lds_off;                                        // lds_addr
  g0[2] = (unsigned)(ga & 0xFFFFFFFFu);                   // global_addr lo
  g0[3] = (unsigned)((ga >> 32) & 0x1FFFFFFu) | (2u << 30); // addr hi | type=2
  i32x8 g1;
  g1[0] = (int)(2u << 16);                                // data_size=4B
  g1[1] = (int)((dim0 & 0xFFFFu) << 16);                  // tensor_dim0[15:0]
  g1[2] = (int)(((dim0 >> 16) & 0xFFFFu) | ((dim1 & 0xFFFFu) << 16));
  g1[3] = (int)(((dim1 >> 16) & 0xFFFFu) | ((dim0 & 0xFFFFu) << 16)); // tile_dim0
  g1[4] = (int)(dim1 & 0xFFFFu);                          // tile_dim1 (tile_dim2=0)
  g1[5] = (int)dim0;                                      // tensor_dim0_stride lo
  g1[6] = 0;
  g1[7] = 0;
  i32x4 z; z[0] = 0; z[1] = 0; z[2] = 0; z[3] = 0;
  i32x8 z8;
#pragma unroll
  for (int i = 0; i < 8; ++i) z8[i] = 0;
  __builtin_amdgcn_tensor_load_to_lds(g0, g1, z, z, z8, 0);
}

// Gather-mode load: 16 rows of a (64-row x 64-col) f32 tensor, 16-bit indices.
__device__ __forceinline__ void tdm_gather_rows(unsigned lds_off, const float* base,
                                                const int* sid16) {
  unsigned long long ga = (unsigned long long)(uintptr_t)base;
  u32x4 g0;
  g0[0] = 1u | (1u << 31);                                // count=1, gather_mode, 16b idx
  g0[1] = lds_off;
  g0[2] = (unsigned)(ga & 0xFFFFFFFFu);
  g0[3] = (unsigned)((ga >> 32) & 0x1FFFFFFu) | (2u << 30);
  i32x8 g1;
  g1[0] = (int)(2u << 16);                                // data_size=4B
  g1[1] = (int)(64u << 16);                               // tensor_dim0 = 64
  g1[2] = (int)(64u << 16);                               // tensor_dim1 = 64 (VOCAB)
  g1[3] = (int)(64u << 16);                               // tile_dim0 = 64
  g1[4] = 16;                                             // tile_dim1 = 16 indices
  g1[5] = 64;                                             // tensor_dim0_stride = 64
  g1[6] = 0;
  g1[7] = 0;
  i32x4 g2, g3;
#pragma unroll
  for (int i = 0; i < 4; ++i)
    g2[i] = (sid16[2 * i] & 0xFFFF) | (sid16[2 * i + 1] << 16);
#pragma unroll
  for (int i = 0; i < 4; ++i)
    g3[i] = (sid16[8 + 2 * i] & 0xFFFF) | (sid16[9 + 2 * i] << 16);
  i32x8 z8;
#pragma unroll
  for (int i = 0; i < 8; ++i) z8[i] = 0;
  __builtin_amdgcn_tensor_load_to_lds(g0, g1, g2, g3, z8, 0);
}

// ---------- fused forward: one block per batch ----------
__global__ __launch_bounds__(256)
void delta_forward(const int* __restrict__ seq, const float* __restrict__ embed,
                   const float* __restrict__ w1, const float* __restrict__ b1,
                   const float* __restrict__ w2, const float* __restrict__ b2,
                   const float* __restrict__ ln_g, const float* __restrict__ ln_b,
                   const float* __restrict__ wk, const float* __restrict__ wv,
                   const float* __restrict__ wq, const float* __restrict__ wrp,
                   const float* __restrict__ brp, const float* __restrict__ wout,
                   const float* __restrict__ bout, float* __restrict__ out)
{
  extern __shared__ char smem[];
  bf16*  w1T  = (bf16*)(smem);                 // 128x64 bf16  16384 B
  bf16*  w2T  = (bf16*)(smem + 16384);         // 64x128 bf16  16384
  bf16*  wkT  = (bf16*)(smem + 32768);         // 64x64  bf16   8192
  bf16*  wvT  = (bf16*)(smem + 40960);         // 64x64  bf16   8192
  float* b1s  = (float*)(smem + 49152);        // 128 f32        512
  float* b2s  = (float*)(smem + 49664);        // 64  f32        256
  float* gs   = (float*)(smem + 49920);        // 64  f32        256
  float* bbs  = (float*)(smem + 50176);        // 64  f32        256
  float* hlast= (float*)(smem + 50432);        // 64  f32        256
  bf16*  ebf  = (bf16*)(smem + 50688);         // 64x64 bf16 (e, then hs)  8192
  bf16*  h1   = (bf16*)(smem + 58880);         // 64x128 bf16  16384
  bf16*  kcb  = (bf16*)(smem + 75264);         // Kc  64x64     8192
  bf16*  kct  = (bf16*)(smem + 83456);         // Kc^T 64x64    8192
  bf16*  vcb  = (bf16*)(smem + 91648);         // Vc  64x64     8192
  bf16*  utb  = (bf16*)(smem + 99840);         // U^T 64x64     8192
  bf16*  mh   = (bf16*)(smem + 108032);        // M (bf16 copy) 8192
  float* Mf   = (float*)(smem + 116224);       // M f32        16384
  float* f32a = (float*)(smem + 132608);       // 64x128 f32   32768
  float* f32b = (float*)(smem + 165376);       // 64x64  f32   16384
  bf16*  abf  = (bf16*)(smem + 181760);        // -Amat bf16    8192
  // total = 189952 bytes

  const int tid = (int)threadIdx.x;
  const int nth = (int)blockDim.x;
  const int b   = (int)blockIdx.x;

  // ---- stage raw f32 weights into LDS via the Tensor Data Mover ----
  // staging region [50688, 148992) is free at startup.
  const unsigned stage = (unsigned)(uintptr_t)(smem + 50688);
  if (tid < 32) {                               // one wave issues the DMAs
    tdm_load_2d(stage,          w1, FDIM, HDIM);   // w1 (64x128)
    tdm_load_2d(stage + 32768,  w2, HDIM, FDIM);   // w2 (128x64)
    tdm_load_2d(stage + 65536,  wk, HDIM, HDIM);
    tdm_load_2d(stage + 81920,  wv, HDIM, HDIM);
    __builtin_amdgcn_s_wait_tensorcnt(0);
  }
  __syncthreads();
  {
    const float* w1f = (const float*)(smem + 50688);
    const float* w2f = (const float*)(smem + 83456);
    const float* wkf = (const float*)(smem + 116224);
    const float* wvf = (const float*)(smem + 132608);
    for (int i = tid; i < HDIM * FDIM; i += nth) {        // w1 (64,128) -> w1T
      int k = i >> 7, n = i & 127;
      w1T[n * HDIM + k] = (bf16)w1f[i];
    }
    for (int i = tid; i < FDIM * HDIM; i += nth) {        // w2 (128,64) -> w2T
      int k = i >> 6, n = i & 63;
      w2T[n * FDIM + k] = (bf16)w2f[i];
    }
    for (int i = tid; i < HDIM * HDIM; i += nth) {
      int k = i >> 6, n = i & 63;
      wkT[n * HDIM + k] = (bf16)wkf[i];
      wvT[n * HDIM + k] = (bf16)wvf[i];
    }
    for (int i = tid; i < FDIM; i += nth) b1s[i] = b1[i];
    for (int i = tid; i < HDIM; i += nth) { b2s[i] = b2[i]; gs[i] = ln_g[i]; bbs[i] = ln_b[i]; }
  }
  __syncthreads();
  for (int i = tid; i < HDIM * HDIM; i += nth) { Mf[i] = 0.f; mh[i] = (bf16)0.f; }
  __syncthreads();

  const int* seqb = seq + b * SEQL;
  const unsigned f32aOff = (unsigned)(uintptr_t)f32a;

  for (int c0 = 0; c0 < SEQL; c0 += CH) {
    if ((c0 + CH < SEQL) && tid < CH)
      __builtin_prefetch(&seqb[c0 + CH + tid], 0, 1);     // global_prefetch_b8

    // ---- embedding gather via TDM gather-mode (16 rows per descriptor) ----
    if (tid < 32) {
      for (int g = 0; g < 4; ++g) {
        int sid[16];
#pragma unroll
        for (int r = 0; r < 16; ++r)
          sid[r] = __builtin_amdgcn_readfirstlane(seqb[c0 + g * 16 + r]);
        tdm_gather_rows(f32aOff + (unsigned)g * 4096u, embed, sid);
      }
      __builtin_amdgcn_s_wait_tensorcnt(0);
    }
    __syncthreads();
    for (int i = tid; i < CH * HDIM; i += nth) ebf[i] = (bf16)f32a[i];
    __syncthreads();

    // ---- FF1: f32a(64x128) = e @ w1 ----
    gemm_lds(f32a, FDIM, ebf, HDIM, w1T, HDIM, CH, FDIM, HDIM, nullptr);
    __syncthreads();
    for (int i = tid; i < CH * FDIM; i += nth) {
      int n = i & 127;
      h1[i] = (bf16)fmaxf(f32a[i] + b1s[n], 0.f);
    }
    __syncthreads();

    // ---- FF2: f32a(64x64) = h1 @ w2 ----
    gemm_lds(f32a, HDIM, h1, FDIM, w2T, FDIM, CH, HDIM, FDIM, nullptr);
    __syncthreads();

    // ---- residual ----
    for (int i = tid; i < CH * HDIM; i += nth) {
      int n = i & 63;
      f32a[i] = f32a[i] + b2s[n] + (float)ebf[i];
    }
    __syncthreads();

    // ---- LayerNorm per row -> hs (bf16, overwrites ebf) ----
    if (tid < CH) {
      const float* xr = f32a + tid * HDIM;
      float mu = 0.f;
#pragma unroll
      for (int h2 = 0; h2 < HDIM; ++h2) mu += xr[h2];
      mu *= (1.f / HDIM);
      float var = 0.f;
#pragma unroll
      for (int h2 = 0; h2 < HDIM; ++h2) { float d = xr[h2] - mu; var += d * d; }
      var *= (1.f / HDIM);
      float rstd = rsqrtf(var + 1e-5f);
      bool isLast = (c0 + tid == SEQL - 1);
#pragma unroll
      for (int h2 = 0; h2 < HDIM; ++h2) {
        float hv = (xr[h2] - mu) * rstd * gs[h2] + bbs[h2];
        ebf[tid * HDIM + h2] = (bf16)hv;
        if (isLast) hlast[h2] = hv;
      }
    }
    __syncthreads();

    // ---- K/V projections ----
    gemm_lds(f32a, HDIM, ebf, HDIM, wkT, HDIM, CH, HDIM, HDIM, nullptr);  // K
    gemm_lds(f32b, HDIM, ebf, HDIM, wvT, HDIM, CH, HDIM, HDIM, nullptr);  // V
    __syncthreads();

    // ---- normalize K rows, pack Kc / Kc^T / Vc (last position zeroed) ----
    if (tid < CH) {
      int t = tid;
      bool drop = (c0 + t == SEQL - 1);                   // hs[:, :-1] only
      float nv = 0.f;
#pragma unroll
      for (int h2 = 0; h2 < HDIM; ++h2) { float kv = f32a[t * HDIM + h2]; nv += kv * kv; }
      float s = drop ? 0.f : (1.f / fmaxf(sqrtf(nv), 1e-12f));
#pragma unroll
      for (int h2 = 0; h2 < HDIM; ++h2) {
        bf16 kv = (bf16)(f32a[t * HDIM + h2] * s);
        kcb[t * HDIM + h2] = kv;
        kct[h2 * HDIM + t] = kv;
        vcb[t * HDIM + h2] = drop ? (bf16)0.f : (bf16)f32b[t * HDIM + h2];
      }
    }
    __syncthreads();

    // ---- chunked delta rule ----
    // P = Kc @ M^T   (Bt of M^T is M row-major -> mh)
    gemm_lds(f32a, HDIM, kcb, HDIM, mh,  HDIM, CH, HDIM, HDIM, nullptr);
    // A = Kc @ Kc^T  (Bt of Kc^T is Kc row-major)
    gemm_lds(f32b, HDIM, kcb, HDIM, kcb, HDIM, CH, CH,   HDIM, nullptr);
    __syncthreads();

    // rhs U = V - P (in place); abf = -A (bf16, for WMMA subtraction); utb = 0
    for (int i = tid; i < CH * HDIM; i += nth) {
      f32a[i] = (float)vcb[i] - f32a[i];
      abf[i]  = (bf16)(-f32b[i]);
      utb[i]  = (bf16)0.f;
    }
    __syncthreads();

    // ---- blocked forward substitution: (I + tril(A,-1)) U = rhs ----
    // 16-row blocks; off-diagonal corrections via WMMA (u_I += (-A[I,:])·U),
    // zero rows of utb make the K padding inert; 16x16 diagonal stays serial.
    for (int I = 0; I < 4; ++I) {
      if (I > 0) {
        int Kpad = (I == 3) ? 64 : 32;
        gemm_lds(f32a + I * 16 * HDIM, HDIM,
                 abf  + I * 16 * CH,   CH,
                 utb, CH, 16, HDIM, Kpad,
                 f32a + I * 16 * HDIM);
        __syncthreads();
      }
      if (tid < HDIM) {
        int j = tid, base = I * 16;
        for (int t = base + 1; t < base + 16; ++t) {
          float acc = f32a[t * HDIM + j];
          for (int s2 = base; s2 < t; ++s2)
            acc -= f32b[t * HDIM + s2] * f32a[s2 * HDIM + j];
          f32a[t * HDIM + j] = acc;
        }
#pragma unroll
        for (int t = base; t < base + 16; ++t)
          utb[j * CH + t] = (bf16)f32a[t * HDIM + j];     // U^T row j = column j of U
      }
      __syncthreads();
    }

    // M += U^T @ Kc   (A = U^T rows; Bt = Kc^T rows; C accumulates from Mf)
    gemm_lds(Mf, HDIM, utb, CH, kct, HDIM, HDIM, HDIM, CH, Mf);
    __syncthreads();
    for (int i = tid; i < HDIM * HDIM; i += nth) mh[i] = (bf16)Mf[i];
    __syncthreads();
  }

  // ---- head: q = hs_last @ wq ; r = M q ; out = (r @ wrp + brp) @ wout + bout ----
  float* qv  = f32a;
  float* rv  = f32a + 64;
  float* rpv = f32a + 128;
  if (tid < HDIM) {
    float q = 0.f;
#pragma unroll
    for (int h2 = 0; h2 < HDIM; ++h2) q += hlast[h2] * wq[h2 * HDIM + tid];
    qv[tid] = q;
  }
  __syncthreads();
  if (tid < HDIM) {
    float r = 0.f;
#pragma unroll
    for (int j2 = 0; j2 < HDIM; ++j2) r += Mf[tid * HDIM + j2] * qv[j2];
    rv[tid] = r;
  }
  __syncthreads();
  if (tid < HDIM) {
    float rp = brp[tid];
#pragma unroll
    for (int i2 = 0; i2 < HDIM; ++i2) rp += rv[i2] * wrp[i2 * HDIM + tid];
    rpv[tid] = rp;
  }
  __syncthreads();
  if (tid < HDIM) {
    float o = bout[tid];
#pragma unroll
    for (int j2 = 0; j2 < HDIM; ++j2) o += rpv[j2] * wout[j2 * HDIM + tid];
    out[b * HDIM + tid] = o;
  }
}

extern "C" void kernel_launch(void* const* d_in, const int* in_sizes, int n_in,
                              void* d_out, int out_size, void* d_ws, size_t ws_size,
                              hipStream_t stream) {
  const int*   seq   = (const int*)  d_in[0];
  const float* embed = (const float*)d_in[1];
  const float* w1    = (const float*)d_in[2];
  const float* b1    = (const float*)d_in[3];
  const float* w2    = (const float*)d_in[4];
  const float* b2    = (const float*)d_in[5];
  const float* ln_g  = (const float*)d_in[6];
  const float* ln_b  = (const float*)d_in[7];
  const float* wk    = (const float*)d_in[8];
  const float* wv    = (const float*)d_in[9];
  const float* wq    = (const float*)d_in[10];
  const float* wrp   = (const float*)d_in[11];
  const float* brp   = (const float*)d_in[12];
  const float* wout  = (const float*)d_in[13];
  const float* bout  = (const float*)d_in[14];
  float* out = (float*)d_out;

  constexpr int SMEM = 189952;   // ~186 KB dynamic LDS (WGP has 320 KB)
  (void)hipFuncSetAttribute(reinterpret_cast<const void*>(delta_forward),
                            hipFuncAttributeMaxDynamicSharedMemorySize, SMEM);
  delta_forward<<<NB, 256, SMEM, stream>>>(seq, embed, w1, b1, w2, b2, ln_g, ln_b,
                                           wk, wv, wq, wrp, brp, wout, bout, out);
}
